// RNNModel_87041807221088
// MI455X (gfx1250) — matrix operand order repaired and checked by reading, not compiled, not run on hardware
//
#include <hip/hip_runtime.h>
#include <hip/hip_bf16.h>

typedef __attribute__((ext_vector_type(16))) _Float16 v16h;
typedef __attribute__((ext_vector_type(8)))  _Float16 v8h;
typedef __attribute__((ext_vector_type(8)))  float    v8f;
typedef __attribute__((ext_vector_type(4)))  unsigned int v4u;
typedef __attribute__((ext_vector_type(8)))  int      v8i;
typedef __attribute__((ext_vector_type(4)))  int      v4i;

#define B_TOT 8192
#define T_LEN 256
#define F_IN  64
#define H_RNN 50
#define WSTR  72      // LDS stride (halfs) for weight rows: 144B, 16B aligned, conflict-free
#define STG   72      // LDS stride (halfs) for h staging rows
#define XSTR  68      // LDS stride (floats) for TDM x tile: 272B, via pad_amount=4 DWORDs
#define NWAVES 4

__device__ __forceinline__ v8f wmma16(v16h a, v16h b, v8f c) {
  return __builtin_amdgcn_wmma_f32_16x16x32_f16(false, a, false, b, (short)0, c, false, false);
}

__device__ __forceinline__ float tanh_fast(float x) {
  // tanh(x) = 1 - 2/(exp(2x)+1); exp(2x) = exp2(x * 2*log2(e))
  float e = __builtin_amdgcn_exp2f(x * 2.8853900817779268f);
  return 1.0f - 2.0f * __builtin_amdgcn_rcpf(e + 1.0f);
}

__device__ __forceinline__ float sigmoid_fast(float x) {
  return __builtin_amdgcn_rcpf(1.0f + __builtin_amdgcn_exp2f(-1.4426950408889634f * x));
}

// B-fragment (32x16 f16, K x N) from LDS weight tile stored row-major [n][k] (stride WSTR).
// Lane holds column n = (lane&15)+16*nt; halfs j<8 = W[n][kb+j], j>=8 = W[n][kb+16+(j-8)].
__device__ __forceinline__ v16h ldsB(const _Float16* w, int lane, int kt, int nt) {
  int n  = (lane & 15) + (nt << 4);
  int kb = ((lane >> 4) << 3) + (kt << 5);   // (lane<16?0:8) + kt*32
  const _Float16* p = w + n * WSTR + kb;
  union { v16h v; v8h h[2]; } u;
  u.h[0] = *(const v8h*)(p);
  u.h[1] = *(const v8h*)(p + 16);
  return u.v;
}

// A-fragment (16x32 f16, M x K) from LDS staging tile row-major [m][k] (stride STG).
__device__ __forceinline__ v16h ldsA(const _Float16* s, int lane, int kt) {
  int m  = lane & 15;
  int kb = ((lane >> 4) << 3) + (kt << 5);   // (lane<16?0:8) + kt*32
  const _Float16* p = s + m * STG + kb;
  union { v16h v; v8h h[2]; } u;
  u.h[0] = *(const v8h*)(p);
  u.h[1] = *(const v8h*)(p + 16);
  return u.v;
}

__device__ __forceinline__ v16h cvt16(float4 a0, float4 a1, float4 b0, float4 b1) {
  v16h r;
  r[0]=(_Float16)a0.x; r[1]=(_Float16)a0.y; r[2]=(_Float16)a0.z; r[3]=(_Float16)a0.w;
  r[4]=(_Float16)a1.x; r[5]=(_Float16)a1.y; r[6]=(_Float16)a1.z; r[7]=(_Float16)a1.w;
  r[8]=(_Float16)b0.x; r[9]=(_Float16)b0.y; r[10]=(_Float16)b0.z; r[11]=(_Float16)b0.w;
  r[12]=(_Float16)b1.x; r[13]=(_Float16)b1.y; r[14]=(_Float16)b1.z; r[15]=(_Float16)b1.w;
  return r;
}

// A-fragment for x_t from the TDM-staged fp32 LDS tile (row stride XSTR floats).
__device__ __forceinline__ v16h ldsXA(const float* sx, int lane, int kt) {
  int m  = lane & 15;
  int kb = ((lane >> 4) << 3) + (kt << 5);
  const float* p = sx + m * XSTR + kb;
  float4 a0 = *(const float4*)(p);
  float4 a1 = *(const float4*)(p + 4);
  float4 b0 = *(const float4*)(p + 16);
  float4 b1 = *(const float4*)(p + 20);
  return cvt16(a0, a1, b0, b1);
}

#if __has_builtin(__builtin_amdgcn_tensor_load_to_lds)
#define USE_TDM 1
// Issue TDM: 2D tile [16 rows x 64 fp32], row stride T*F elements, into LDS at
// lds_off with 4-DWORD padding every 64 DWORDs (-> row stride 68 floats).
__device__ __forceinline__ void tdm_load_x(unsigned int lds_off, const float* gp) {
  unsigned long long ga = (unsigned long long)(uintptr_t)gp;
  v4u g0;
  g0[0] = 1u;                                        // count=1, user descriptor
  g0[1] = lds_off;                                   // lds_addr (bytes)
  g0[2] = (unsigned int)ga;                          // global_addr[31:0]
  g0[3] = (unsigned int)((ga >> 32) & 0x1FFFFFFu)    // global_addr[56:32]
          | (2u << 30);                              // type=2 ("image")
  v8i g1;
  g1[0] = (int)((2u << 16)      // data_size = 4B
              | (1u << 20)      // pad_enable
              | (5u << 22)      // pad_interval: 64 DWORDs (=256B row)
              | (3u << 25));    // pad_amount: 4 DWORDs (row stride -> 272B)
  g1[1] = (int)(64u << 16);     // tensor_dim0 = 64 (low 16 bits at bit 48)
  g1[2] = (int)(16u << 16);     // tensor_dim1 = 16 (low 16 bits at bit 80)
  g1[3] = (int)(64u << 16);     // tile_dim0 = 64
  g1[4] = (int)16;              // tile_dim1 = 16, tile_dim2 = 0
  g1[5] = (int)(T_LEN * F_IN);  // tensor_dim0_stride = 16384 elements
  g1[6] = 0;
  g1[7] = 0;
  v4i z4 = {0, 0, 0, 0};
  v8i z8 = {0, 0, 0, 0, 0, 0, 0, 0};
  // 6-arg toolchain variant: (v4u g0, v8i g1, v4i g2, v4i g3, v8i g4, i32 cpol)
  __builtin_amdgcn_tensor_load_to_lds(g0, g1, z4, z4, z8, 0);
}
#else
#define USE_TDM 0
#endif

__global__ __launch_bounds__(NWAVES * 32)
void rnn2_fused_kernel(const float* __restrict__ x,
                       const float* __restrict__ wih1, const float* __restrict__ whh1,
                       const float* __restrict__ bih1, const float* __restrict__ bhh1,
                       const float* __restrict__ wih2, const float* __restrict__ whh2,
                       const float* __restrict__ bih2, const float* __restrict__ bhh2,
                       const float* __restrict__ fcw,  const float* __restrict__ fcb,
                       float* __restrict__ out) {
  __shared__ _Float16 s_wih1[64 * WSTR];
  __shared__ _Float16 s_whh1[64 * WSTR];
  __shared__ _Float16 s_wih2[64 * WSTR];
  __shared__ _Float16 s_whh2[64 * WSTR];
  __shared__ float    s_b1[64];
  __shared__ float    s_b2[64];
  __shared__ float    s_fcw[64];
  __shared__ _Float16 s_stage[NWAVES * 16 * STG];
  __shared__ float    s_x[NWAVES * 16 * XSTR];       // TDM destination, per-wave tile

  const int tid  = threadIdx.x;
  const int bs   = blockDim.x;
  const int lane = tid & 31;
  const int wave = tid >> 5;

  // ---- cooperative weight convert (fp32 -> f16, zero-padded to 64x64) ----
  for (int i = tid; i < 64 * WSTR; i += bs) {
    s_wih1[i] = (_Float16)0.0f; s_whh1[i] = (_Float16)0.0f;
    s_wih2[i] = (_Float16)0.0f; s_whh2[i] = (_Float16)0.0f;
  }
  __syncthreads();
  for (int i = tid; i < H_RNN * F_IN; i += bs) {       // w_ih1: [50][64]
    int n = i >> 6, k = i & 63;
    s_wih1[n * WSTR + k] = (_Float16)wih1[i];
  }
  for (int i = tid; i < H_RNN * H_RNN; i += bs) {      // 50x50 matrices
    int n = i / H_RNN, k = i - n * H_RNN;
    s_whh1[n * WSTR + k] = (_Float16)whh1[i];
    s_wih2[n * WSTR + k] = (_Float16)wih2[i];
    s_whh2[n * WSTR + k] = (_Float16)whh2[i];
  }
  for (int i = tid; i < 64; i += bs) {
    s_b1[i]  = (i < H_RNN) ? (bih1[i] + bhh1[i]) : 0.0f;
    s_b2[i]  = (i < H_RNN) ? (bih2[i] + bhh2[i]) : 0.0f;
    s_fcw[i] = (i < H_RNN) ? fcw[i] : 0.0f;
  }
  __syncthreads();

  // ---- per-wave setup ----
  const int mb = (blockIdx.x * NWAVES + wave) << 4;    // 16 batch rows per wave
  _Float16* sw = s_stage + wave * 16 * STG;            // private h staging tile
  float*    sx = s_x     + wave * 16 * XSTR;           // private x tile (TDM dest)
  const float* gxbase = x + (size_t)mb * T_LEN * F_IN; // tile start at t=0

  float biasv1[4], biasv2[4], fcwv[4];
#pragma unroll
  for (int nt = 0; nt < 4; ++nt) {
    int n = (lane & 15) + (nt << 4);
    biasv1[nt] = s_b1[n];
    biasv2[nt] = s_b2[n];
    fcwv[nt]   = s_fcw[n];
  }

#if USE_TDM
  const unsigned int sx_off = (unsigned int)(uintptr_t)(void*)sx;  // LDS byte offset (addr[31:0])
  tdm_load_x(sx_off, gxbase);                                      // preload t=0
#endif

  v16h h1a[2] = {}, h2a[2] = {};
  v8f  acc[4];
  v16h b1v[2][4], b2v[2][4];                           // resident B-fragments per layer

  for (int t = 0; t < T_LEN; ++t) {
#if USE_TDM
    asm volatile("s_wait_tensorcnt 0" ::: "memory");   // x tile for t is in LDS
    v16h xa0 = ldsXA(sx, lane, 0);
    v16h xa1 = ldsXA(sx, lane, 1);
    asm volatile("s_wait_dscnt 0" ::: "memory");       // sx reads done before TDM overwrite
    int tn = (t < T_LEN - 1) ? t + 1 : t;              // uniform; last issue is harmless reload
    tdm_load_x(sx_off, gxbase + tn * F_IN);            // async prefetch of next timestep
#else
    const float* px = gxbase + (size_t)(lane & 15) * T_LEN * F_IN + t * F_IN;
    int kb0 = ((lane >> 4) << 3);
    v16h xa0 = cvt16(*(const float4*)(px + kb0),      *(const float4*)(px + kb0 + 4),
                     *(const float4*)(px + kb0 + 16), *(const float4*)(px + kb0 + 20));
    v16h xa1 = cvt16(*(const float4*)(px + kb0 + 32), *(const float4*)(px + kb0 + 36),
                     *(const float4*)(px + kb0 + 48), *(const float4*)(px + kb0 + 52));
#endif

    // ---- batch-load all layer-1 B fragments (32 ds_load_b128, pipelined) ----
#pragma unroll
    for (int nt = 0; nt < 4; ++nt) {
      b1v[0][nt] = ldsB(s_wih1, lane, 0, nt);
      b1v[1][nt] = ldsB(s_wih1, lane, 1, nt);
    }
#pragma unroll
    for (int nt = 0; nt < 4; ++nt) {
      b2v[0][nt] = ldsB(s_whh1, lane, 0, nt);
      b2v[1][nt] = ldsB(s_whh1, lane, 1, nt);
    }

    // ---- layer 1: acc = bias + x W_ih1^T + h1 W_hh1^T ----
#pragma unroll
    for (int nt = 0; nt < 4; ++nt) {
      v8f c;
#pragma unroll
      for (int j = 0; j < 8; ++j) c[j] = biasv1[nt];
      c = wmma16(xa0,    b1v[0][nt], c);
      c = wmma16(xa1,    b1v[1][nt], c);
      c = wmma16(h1a[0], b2v[0][nt], c);
      c = wmma16(h1a[1], b2v[1][nt], c);
      acc[nt] = c;
    }

    // ---- issue layer-2 B fragment loads now: in flight under tanh/transpose ----
#pragma unroll
    for (int nt = 0; nt < 4; ++nt) {
      b1v[0][nt] = ldsB(s_wih2, lane, 0, nt);
      b1v[1][nt] = ldsB(s_wih2, lane, 1, nt);
    }
#pragma unroll
    for (int nt = 0; nt < 4; ++nt) {
      b2v[0][nt] = ldsB(s_whh2, lane, 0, nt);
      b2v[1][nt] = ldsB(s_whh2, lane, 1, nt);
    }

    // tanh + transpose through LDS (D-layout -> A-layout)
#pragma unroll
    for (int nt = 0; nt < 4; ++nt) {
      int n = (lane & 15) + (nt << 4);
#pragma unroll
      for (int j = 0; j < 8; ++j) {
        float th = tanh_fast(acc[nt][j]);
        int m = ((lane >> 4) << 3) + j;
        sw[m * STG + n] = (_Float16)th;
      }
    }
    asm volatile("s_wait_dscnt 0" ::: "memory");       // cross-lane LDS RAW
    h1a[0] = ldsA(sw, lane, 0);
    h1a[1] = ldsA(sw, lane, 1);

    // ---- layer 2: acc = bias + h1 W_ih2^T + h2 W_hh2^T ----
#pragma unroll
    for (int nt = 0; nt < 4; ++nt) {
      v8f c;
#pragma unroll
      for (int j = 0; j < 8; ++j) c[j] = biasv2[nt];
      c = wmma16(h1a[0], b1v[0][nt], c);
      c = wmma16(h1a[1], b1v[1][nt], c);
      c = wmma16(h2a[0], b2v[0][nt], c);
      c = wmma16(h2a[1], b2v[1][nt], c);
      acc[nt] = c;
    }
#pragma unroll
    for (int nt = 0; nt < 4; ++nt) {
      int n = (lane & 15) + (nt << 4);
#pragma unroll
      for (int j = 0; j < 8; ++j) {
        float th = tanh_fast(acc[nt][j]);
        acc[nt][j] = th;                               // keep f32 h2 for final FC
        int m = ((lane >> 4) << 3) + j;
        sw[m * STG + n] = (_Float16)th;
      }
    }
    asm volatile("s_wait_dscnt 0" ::: "memory");
    h2a[0] = ldsA(sw, lane, 0);
    h2a[1] = ldsA(sw, lane, 1);
  }

  // ---- final: relu(h2_T) . fc_w + fc_b -> sigmoid ----
  float fcb0 = fcb[0];
#pragma unroll
  for (int j = 0; j < 8; ++j) {
    float p = 0.0f;
#pragma unroll
    for (int nt = 0; nt < 4; ++nt)
      p += fmaxf(acc[nt][j], 0.0f) * fcwv[nt];
    p += __shfl_xor(p, 1, 32);
    p += __shfl_xor(p, 2, 32);
    p += __shfl_xor(p, 4, 32);
    p += __shfl_xor(p, 8, 32);
    if ((lane & 15) == 0) {
      int m = ((lane >> 4) << 3) + j;                  // rows 0..7 (lane 0), 8..15 (lane 16)
      out[mb + m] = sigmoid_fast(p + fcb0);
    }
  }
}

extern "C" void kernel_launch(void* const* d_in, const int* in_sizes, int n_in,
                              void* d_out, int out_size, void* d_ws, size_t ws_size,
                              hipStream_t stream) {
  const float* x    = (const float*)d_in[0];
  const float* wih1 = (const float*)d_in[1];
  const float* whh1 = (const float*)d_in[2];
  const float* bih1 = (const float*)d_in[3];
  const float* bhh1 = (const float*)d_in[4];
  const float* wih2 = (const float*)d_in[5];
  const float* whh2 = (const float*)d_in[6];
  const float* bih2 = (const float*)d_in[7];
  const float* bhh2 = (const float*)d_in[8];
  const float* fcw  = (const float*)d_in[9];
  const float* fcb  = (const float*)d_in[10];
  float* out = (float*)d_out;

  dim3 grid(B_TOT / (NWAVES * 16));   // 128 workgroups
  dim3 block(NWAVES * 32);            // 128 threads = 4 wave32
  hipLaunchKernelGGL(rnn2_fused_kernel, grid, block, 0, stream,
                     x, wih1, whh1, bih1, bhh1, wih2, whh2, bih2, bhh2, fcw, fcb, out);
}
